// PointTransformerSeg_78271484002540
// MI455X (gfx1250) — matrix operand order repaired and checked by reading, not compiled, not run on hardware
//
#include <hip/hip_runtime.h>
#include <math.h>

// ---------------------------------------------------------------------------
// CDNA5 WMMA types
// ---------------------------------------------------------------------------
typedef __attribute__((ext_vector_type(16))) __bf16 v16bf;
typedef __attribute__((ext_vector_type(8)))  float  v8f;

static __device__ __forceinline__ unsigned short f2bf(float f) {
  unsigned int u = __builtin_bit_cast(unsigned int, f);
  unsigned int r = u + 0x7FFFu + ((u >> 16) & 1u);   // round-to-nearest-even
  return (unsigned short)(r >> 16);
}
static __device__ __forceinline__ __bf16 bfbits(unsigned short u) {
  return __builtin_bit_cast(__bf16, u);
}

// epilogue helper: store one 16x16 f32 C/D tile (per-ISA lane layout)
static __device__ __forceinline__ void pt_store_tile(const v8f& acc, int mbase, int nbase,
                                                     const float* __restrict__ bias,
                                                     const float* __restrict__ resid,
                                                     float* __restrict__ Y,
                                                     int M, int N, int relu, int lane)
{
  const int half = lane >> 4;
  const int col  = lane & 15;
  const int n    = nbase + col;
  const float bval = (bias && n < N) ? bias[n] : 0.0f;
#pragma unroll
  for (int r = 0; r < 8; ++r) {
    const int m = mbase + r + half * 8;
    if (m < M && n < N) {
      float v = acc[r] + bval;
      if (resid) v += resid[(long long)m * N + n];
      if (relu)  v = fmaxf(v, 0.0f);
      Y[(long long)m * N + n] = v;
    }
  }
}

// ---------------------------------------------------------------------------
// Generic GEMM:  Y[M,N] = X[M,K] @ W[N,K]^T (+bias[N]) (+resid[M,N]) (ReLU)
// 128 threads (4 waves) per 64x64 output tile; each wave owns a 32x32 quadrant
// computed as 2x2 sub-tiles of v_wmma_f32_16x16x32_bf16 (4 WMMAs / K-step).
// fp32 inputs staged through LDS as bf16; fp32 accumulation.
// ---------------------------------------------------------------------------
__global__ __launch_bounds__(128)
void pt_gemm_wmma_kernel(const float* __restrict__ X, const float* __restrict__ W,
                         const float* __restrict__ bias, const float* __restrict__ resid,
                         float* __restrict__ Y, int M, int N, int K, int relu)
{
  __shared__ unsigned short As[64][32];   // A tile: rows m0..m0+63, k0..k0+31
  __shared__ unsigned short Bs[64][32];   // W tile: rows n0..n0+63, k0..k0+31

  const int m0   = blockIdx.x * 64;
  const int n0   = blockIdx.y * 64;
  const int tid  = threadIdx.x;
  const int lane = tid & 31;
  const int wave = tid >> 5;
  const int mw   = (wave >> 1) * 32;      // wave quadrant inside 64x64 tile
  const int nw   = (wave & 1) * 32;

  v8f c00 = {0.f,0.f,0.f,0.f,0.f,0.f,0.f,0.f};
  v8f c01 = c00, c10 = c00, c11 = c00;

  for (int k0 = 0; k0 < K; k0 += 32) {
    // speculative prefetch of the next K slab (global_prefetch_b8)
    if (k0 + 32 < K) {
      const int pr = tid >> 1;
      if ((tid & 1) == 0) {
        const int m = m0 + pr;
        if (m < M) __builtin_prefetch(&X[(long long)m * K + k0 + 32], 0, 3);
      } else {
        const int n = n0 + pr;
        if (n < N) __builtin_prefetch(&W[(long long)n * K + k0 + 32], 0, 3);
      }
    }
    // cooperative, coalesced load of both 64x32 tiles (2048 elems each)
    for (int t = tid; t < 64 * 32; t += 128) {
      const int r = t >> 5, cc = t & 31;
      const int m = m0 + r, n = n0 + r, k = k0 + cc;
      As[r][cc] = f2bf((m < M && k < K) ? X[(long long)m * K + k] : 0.0f);
      Bs[r][cc] = f2bf((n < N && k < K) ? W[(long long)n * K + k] : 0.0f);
    }
    __syncthreads();

    const int half = lane >> 4;     // 0: lanes 0-15, 1: lanes 16-31
    const int rc   = lane & 15;
    v16bf a0, a1, b0, b1;
#pragma unroll
    for (int vv = 0; vv < 8; ++vv) {
      // A 16x32 bf16 lane layout: VGPR v holds K pair (v&3)*2 + (v>>2)*16 + half*8
      const int ka = ((vv & 3) << 1) + ((vv >> 2) << 4) + (half << 3);
      a0[2 * vv]     = bfbits(As[mw + rc][ka]);
      a0[2 * vv + 1] = bfbits(As[mw + rc][ka + 1]);
      a1[2 * vv]     = bfbits(As[mw + 16 + rc][ka]);
      a1[2 * vv + 1] = bfbits(As[mw + 16 + rc][ka + 1]);
      // B 32x16 bf16 lane layout: same K order as A (pair (v&3)*2 + (v>>2)*16 + half*8)
      const int kb = ka;
      b0[2 * vv]     = bfbits(Bs[nw + rc][kb]);
      b0[2 * vv + 1] = bfbits(Bs[nw + rc][kb + 1]);
      b1[2 * vv]     = bfbits(Bs[nw + 16 + rc][kb]);
      b1[2 * vv + 1] = bfbits(Bs[nw + 16 + rc][kb + 1]);
    }
    c00 = __builtin_amdgcn_wmma_f32_16x16x32_bf16(false, a0, false, b0, (short)0, c00, false, false);
    c01 = __builtin_amdgcn_wmma_f32_16x16x32_bf16(false, a0, false, b1, (short)0, c01, false, false);
    c10 = __builtin_amdgcn_wmma_f32_16x16x32_bf16(false, a1, false, b0, (short)0, c10, false, false);
    c11 = __builtin_amdgcn_wmma_f32_16x16x32_bf16(false, a1, false, b1, (short)0, c11, false, false);
    __syncthreads();
  }

  pt_store_tile(c00, m0 + mw,      n0 + nw,      bias, resid, Y, M, N, relu, lane);
  pt_store_tile(c01, m0 + mw,      n0 + nw + 16, bias, resid, Y, M, N, relu, lane);
  pt_store_tile(c10, m0 + mw + 16, n0 + nw,      bias, resid, Y, M, N, relu, lane);
  pt_store_tile(c11, m0 + mw + 16, n0 + nw + 16, bias, resid, Y, M, N, relu, lane);
}

// ---------------------------------------------------------------------------
// KNN: per query point, brute-force scan of refs keeping KK smallest sq-dists
// (stable tie order to match lax.top_k on -dist)
// ---------------------------------------------------------------------------
template <int KK>
__global__ void pt_knn_kernel(const float* __restrict__ q, const float* __restrict__ ref,
                              int* __restrict__ outIdx, float* __restrict__ outD,
                              int B, int Nq, int Nr)
{
  const int t = blockIdx.x * blockDim.x + threadIdx.x;
  if (t >= B * Nq) return;
  const int b = t / Nq;
  const float qx = q[t * 3 + 0], qy = q[t * 3 + 1], qz = q[t * 3 + 2];
  const float* R = ref + (long long)b * Nr * 3;

  float bd[KK]; int bi[KK];
#pragma unroll
  for (int i = 0; i < KK; ++i) { bd[i] = 3.0e38f; bi[i] = 0; }

  for (int j = 0; j < Nr; ++j) {
    const float dx = qx - R[j * 3 + 0];
    const float dy = qy - R[j * 3 + 1];
    const float dz = qz - R[j * 3 + 2];
    const float d = dx * dx + dy * dy + dz * dz;
    if (d < bd[KK - 1]) {
      int p = KK - 1;
#pragma unroll
      for (int s = KK - 1; s > 0; --s) {
        if (bd[s - 1] > d) { bd[s] = bd[s - 1]; bi[s] = bi[s - 1]; p = s - 1; }
      }
      bd[p] = d; bi[p] = j;
    }
  }
#pragma unroll
  for (int i = 0; i < KK; ++i) {
    outIdx[(long long)t * KK + i] = bi[i];
    if (outD) outD[(long long)t * KK + i] = bd[i];
  }
}

// ---------------------------------------------------------------------------
// Farthest point sampling: one workgroup per batch, sequential scan
// ---------------------------------------------------------------------------
__global__ __launch_bounds__(256)
void pt_fps_kernel(const float* __restrict__ xyz, int* __restrict__ out,
                   int N, int npoint)
{
  __shared__ float dmin[4096];
  __shared__ float redV[256];
  __shared__ int   redI[256];
  const int b = blockIdx.x, tid = threadIdx.x;
  const float* P = xyz + (long long)b * N * 3;

  for (int i = tid; i < N; i += 256) dmin[i] = 1.0e10f;
  __syncthreads();

  int far = 0;
  for (int it = 0; it < npoint; ++it) {
    if (tid == 0) out[(long long)b * npoint + it] = far;
    const float cx = P[far * 3 + 0], cy = P[far * 3 + 1], cz = P[far * 3 + 2];
    float best = -1.0f; int bestI = N;
    for (int i = tid; i < N; i += 256) {
      const float dx = P[i * 3 + 0] - cx;
      const float dy = P[i * 3 + 1] - cy;
      const float dz = P[i * 3 + 2] - cz;
      const float dm = fminf(dmin[i], dx * dx + dy * dy + dz * dz);
      dmin[i] = dm;
      if (dm > best) { best = dm; bestI = i; }   // strict > keeps lowest index
    }
    redV[tid] = best; redI[tid] = bestI;
    __syncthreads();
    for (int s = 128; s > 0; s >>= 1) {
      if (tid < s) {
        if (redV[tid + s] > redV[tid] ||
            (redV[tid + s] == redV[tid] && redI[tid + s] < redI[tid])) {
          redV[tid] = redV[tid + s]; redI[tid] = redI[tid + s];
        }
      }
      __syncthreads();
    }
    far = redI[0];
    __syncthreads();
  }
}

// ---------------------------------------------------------------------------
// Elementwise / gather kernels
// ---------------------------------------------------------------------------
__global__ void pt_slice3_kernel(const float* __restrict__ x, float* __restrict__ xyz, int total)
{
  const int t = blockIdx.x * blockDim.x + threadIdx.x;
  if (t >= total) return;
  xyz[t] = x[(t / 3) * 6 + (t % 3)];
}

__global__ void pt_gather3_kernel(const float* __restrict__ xyz, const int* __restrict__ fidx,
                                  float* __restrict__ out, int M, int N, int total)
{
  const int t = blockIdx.x * blockDim.x + threadIdx.x;
  if (t >= total) return;
  const int c = t % 3, bm = t / 3, b = bm / M;
  out[t] = xyz[((long long)b * N + fidx[bm]) * 3 + c];
}

// transition_down grouping: out[b,m,k,:] = concat(xyz[idx]-newxyz, feats[idx])
__global__ void pt_tdgroup_kernel(const float* __restrict__ xyz, const float* __restrict__ nxyz,
                                  const float* __restrict__ feats, const int* __restrict__ idx,
                                  float* __restrict__ out, int M, int N, int Cin, int total)
{
  const int t = blockIdx.x * blockDim.x + threadIdx.x;
  if (t >= total) return;
  const int CT = Cin + 3;
  const int ch = t % CT;
  int r = t / CT;
  const int k = r & 15; r >>= 4;
  const int m = r % M, b = r / M;
  const int j = idx[((long long)b * M + m) * 16 + k];
  float v;
  if (ch < 3) v = xyz[((long long)b * N + j) * 3 + ch] - nxyz[((long long)b * M + m) * 3 + ch];
  else        v = feats[((long long)b * N + j) * Cin + (ch - 3)];
  out[t] = v;
}

// rel[rr,k,:] = xyz[row] - xyz[neighbor]   (rows are global b*N+n, chunk [r0,r0+R))
__global__ void pt_relpos_kernel(const float* __restrict__ xyz, const int* __restrict__ idx,
                                 float* __restrict__ rel, int r0, int R, int N)
{
  const int t = blockIdx.x * blockDim.x + threadIdx.x;
  if (t >= R * 16 * 3) return;
  const int c = t % 3, k = (t / 3) & 15, rr = t / 48;
  const int row = r0 + rr, b = row / N;
  const int j = idx[(long long)row * 16 + k];
  rel[t] = xyz[row * 3 + c] - xyz[((long long)b * N + j) * 3 + c];
}

__global__ void pt_gatherf_kernel(const float* __restrict__ src, const int* __restrict__ idx,
                                  float* __restrict__ out, int r0, int R, int N, int F)
{
  const int t = blockIdx.x * blockDim.x + threadIdx.x;
  if (t >= R * 16 * F) return;
  const int f = t % F, k = (t / F) & 15, rr = t / (16 * F);
  const int row = r0 + rr, b = row / N;
  const int j = idx[(long long)row * 16 + k];
  out[t] = src[((long long)b * N + j) * F + f];
}

// attn_in = q(broadcast over k) - kf + pos
__global__ void pt_qsub_kernel(const float* __restrict__ q, const float* __restrict__ kf,
                               const float* __restrict__ pos, float* __restrict__ out,
                               int r0, int R, int F)
{
  const int t = blockIdx.x * blockDim.x + threadIdx.x;
  if (t >= R * 16 * F) return;
  const int f = t % F, rr = t / (16 * F);
  out[t] = q[(long long)(r0 + rr) * F + f] - kf[t] + pos[t];
}

// softmax over the k axis (16) of [R,16,F], with scale applied first
__global__ void pt_softmaxk_kernel(float* __restrict__ attn, int R, int F, float scale)
{
  const int t = blockIdx.x * blockDim.x + threadIdx.x;
  if (t >= R * F) return;
  const int f = t % F, rr = t / F;
  float* p = attn + (long long)rr * 16 * F + f;
  float mx = -3.0e38f;
#pragma unroll
  for (int k = 0; k < 16; ++k) mx = fmaxf(mx, p[k * F] * scale);
  float s = 0.0f;
#pragma unroll
  for (int k = 0; k < 16; ++k) { const float e = expf(p[k * F] * scale - mx); p[k * F] = e; s += e; }
  const float inv = 1.0f / s;
#pragma unroll
  for (int k = 0; k < 16; ++k) p[k * F] *= inv;
}

// res[row,f] = sum_k attn[rr,k,f] * (v[rr,k,f] + pos[rr,k,f])
__global__ void pt_wsum_kernel(const float* __restrict__ attn, const float* __restrict__ v,
                               const float* __restrict__ pos, float* __restrict__ res,
                               int r0, int R, int F)
{
  const int t = blockIdx.x * blockDim.x + threadIdx.x;
  if (t >= R * F) return;
  const int f = t % F, rr = t / F;
  const long long base = (long long)rr * 16 * F + f;
  float s = 0.0f;
#pragma unroll
  for (int k = 0; k < 16; ++k) s += attn[base + k * F] * (v[base + k * F] + pos[base + k * F]);
  res[(long long)(r0 + rr) * F + f] = s;
}

// batchnorm: per-channel mean/var over all rows (biased var, matches jnp.var)
__global__ __launch_bounds__(256)
void pt_bnstats_kernel(const float* __restrict__ x, float* __restrict__ mean,
                       float* __restrict__ var, int rows, int C)
{
  __shared__ float S[256], S2[256];
  const int c = blockIdx.x, tid = threadIdx.x;
  float s = 0.0f, s2 = 0.0f;
  for (int r = tid; r < rows; r += 256) {
    const float v = x[(long long)r * C + c];
    s += v; s2 += v * v;
  }
  S[tid] = s; S2[tid] = s2;
  __syncthreads();
  for (int st = 128; st > 0; st >>= 1) {
    if (tid < st) { S[tid] += S[tid + st]; S2[tid] += S2[tid + st]; }
    __syncthreads();
  }
  if (tid == 0) {
    const float m = S[0] / (float)rows;
    mean[c] = m;
    var[c] = S2[0] / (float)rows - m * m;
  }
}

__global__ void pt_bnapply_kernel(float* __restrict__ x, const float* __restrict__ mean,
                                  const float* __restrict__ var, const float* __restrict__ g,
                                  const float* __restrict__ be, int C, int total, int relu)
{
  const int t = blockIdx.x * blockDim.x + threadIdx.x;
  if (t >= total) return;
  const int c = t % C;
  float v = (x[t] - mean[c]) * rsqrtf(var[c] + 1e-5f) * g[c] + be[c];
  if (relu) v = fmaxf(v, 0.0f);
  x[t] = v;
}

// max over neighbor axis: out[bm,c] = max_k h[bm,k,c]
__global__ void pt_maxk_kernel(const float* __restrict__ h, float* __restrict__ out,
                               int C, int total)
{
  const int t = blockIdx.x * blockDim.x + threadIdx.x;
  if (t >= total) return;
  const int c = t % C, bm = t / C;
  const float* p = h + (long long)bm * 16 * C + c;
  float m = p[0];
#pragma unroll
  for (int k = 1; k < 16; ++k) m = fmaxf(m, p[k * C]);
  out[t] = m;
}

// 3-NN inverse-distance interpolation + add f2
__global__ void pt_interp_kernel(const float* __restrict__ f1, const int* __restrict__ i3,
                                 const float* __restrict__ d3, const float* __restrict__ f2,
                                 float* __restrict__ out, int N2, int N1, int C, int total)
{
  const int t = blockIdx.x * blockDim.x + threadIdx.x;
  if (t >= total) return;
  const int c = t % C, bn = t / C, b = bn / N2;
  const float w0 = 1.0f / (d3[bn * 3 + 0] + 1e-8f);
  const float w1 = 1.0f / (d3[bn * 3 + 1] + 1e-8f);
  const float w2 = 1.0f / (d3[bn * 3 + 2] + 1e-8f);
  const float inv = 1.0f / (w0 + w1 + w2);
  const long long base = (long long)b * N1;
  const float v = w0 * f1[(base + i3[bn * 3 + 0]) * C + c]
                + w1 * f1[(base + i3[bn * 3 + 1]) * C + c]
                + w2 * f1[(base + i3[bn * 3 + 2]) * C + c];
  out[t] = v * inv + f2[t];
}

// ---------------------------------------------------------------------------
// Host-side orchestration
// ---------------------------------------------------------------------------
namespace {

struct Wsp {
  char* base; size_t off; size_t cap;
  void* get(size_t bytes) {
    size_t a = (off + 255) & ~(size_t)255;
    off = a + bytes;
    return base + a;
  }
  float* f(size_t n) { return (float*)get(n * sizeof(float)); }
  int*   i(size_t n) { return (int*)get(n * sizeof(int)); }
  size_t mark() const { return off; }
  void   reset(size_t m) { off = m; }
};

struct Lin { const float* w; const float* b; };
struct BNp { const float* g; const float* be; };
struct TB  { Lin fc1, fc2, d0, d1, g0, g1, wq, wk, wv; };
struct TDp { Lin c0, c1; BNp b0, b1; };
struct TUp { Lin f1; BNp b1; Lin f2; BNp b2; };

struct Cursor {
  void* const* in; int cur;
  const float* next() { return (const float*)in[cur++]; }
};
static Lin parse_lin(Cursor& c, bool bias = true) {
  Lin l; l.w = c.next(); l.b = bias ? c.next() : nullptr; return l;
}
static BNp parse_bn(Cursor& c) { BNp b; b.g = c.next(); b.be = c.next(); return b; }
static TB parse_tb(Cursor& c) {
  TB t;
  t.fc1 = parse_lin(c); t.fc2 = parse_lin(c);
  t.d0 = parse_lin(c); t.d1 = parse_lin(c);
  t.g0 = parse_lin(c); t.g1 = parse_lin(c);
  t.wq = parse_lin(c, false); t.wk = parse_lin(c, false); t.wv = parse_lin(c, false);
  return t;
}

static inline int cdiv(long long a, int b) { return (int)((a + b - 1) / b); }

static void gemm(hipStream_t s, const float* X, const float* W, const float* bias,
                 const float* resid, float* Y, int M, int N, int K, bool relu) {
  dim3 g(cdiv(M, 64), cdiv(N, 64));
  pt_gemm_wmma_kernel<<<g, 128, 0, s>>>(X, W, bias, resid, Y, M, N, K, relu ? 1 : 0);
}

static void bn_relu(hipStream_t s, Wsp& ws, float* x, const BNp& p, int rows, int C, bool relu) {
  float* mv = ws.f(2 * (size_t)C);
  pt_bnstats_kernel<<<C, 256, 0, s>>>(x, mv, mv + C, rows, C);
  const int total = rows * C;
  pt_bnapply_kernel<<<cdiv(total, 256), 256, 0, s>>>(x, mv, mv + C, p.g, p.be, C, total, relu ? 1 : 0);
}

static const float SMSCALE = 0.04419417382415922f; // 1/sqrt(512)

static void run_transformer(hipStream_t s, Wsp& ws, const TB& tb,
                            const float* xyz, float* feats, int B, int N, int C) {
  const int DM = 512;
  const size_t mk = ws.mark();
  const int R = B * N;

  int* idx = ws.i((size_t)R * 16);
  pt_knn_kernel<16><<<cdiv(R, 128), 128, 0, s>>>(xyz, xyz, idx, nullptr, B, N, N);

  float* x  = ws.f((size_t)R * DM);
  float* q  = ws.f((size_t)R * DM);
  float* kf = ws.f((size_t)R * DM);
  float* vf = ws.f((size_t)R * DM);
  float* rs = ws.f((size_t)R * DM);
  gemm(s, feats, tb.fc1.w, tb.fc1.b, nullptr, x, R, DM, C, false);
  gemm(s, x, tb.wq.w, nullptr, nullptr, q,  R, DM, DM, false);
  gemm(s, x, tb.wk.w, nullptr, nullptr, kf, R, DM, DM, false);
  gemm(s, x, tb.wv.w, nullptr, nullptr, vf, R, DM, DM, false);

  int CR = 512; if (CR > R) CR = R;
  float* rel = ws.f((size_t)CR * 16 * 3);
  float* c1 = ws.f((size_t)CR * 16 * DM);
  float* c2 = ws.f((size_t)CR * 16 * DM);
  float* c3 = ws.f((size_t)CR * 16 * DM);
  float* c4 = ws.f((size_t)CR * 16 * DM);

  for (int r0 = 0; r0 < R; r0 += CR) {
    const int Rc = (R - r0 < CR) ? (R - r0) : CR;
    const int rows = Rc * 16;
    pt_relpos_kernel<<<cdiv(rows * 3, 256), 256, 0, s>>>(xyz, idx, rel, r0, Rc, N);
    gemm(s, rel, tb.d0.w, tb.d0.b, nullptr, c2, rows, DM, 3, true);     // delta hidden
    gemm(s, c2, tb.d1.w, tb.d1.b, nullptr, c1, rows, DM, DM, false);    // pos -> c1
    pt_gatherf_kernel<<<cdiv(rows * DM, 256), 256, 0, s>>>(kf, idx, c3, r0, Rc, N, DM);
    pt_qsub_kernel<<<cdiv(rows * DM, 256), 256, 0, s>>>(q, c3, c1, c2, r0, Rc, DM); // attn_in -> c2
    gemm(s, c2, tb.g0.w, tb.g0.b, nullptr, c3, rows, DM, DM, true);     // gamma hidden
    gemm(s, c3, tb.g1.w, tb.g1.b, nullptr, c2, rows, DM, DM, false);    // attn -> c2
    pt_softmaxk_kernel<<<cdiv(Rc * DM, 256), 256, 0, s>>>(c2, Rc, DM, SMSCALE);
    pt_gatherf_kernel<<<cdiv(rows * DM, 256), 256, 0, s>>>(vf, idx, c4, r0, Rc, N, DM);
    pt_wsum_kernel<<<cdiv(Rc * DM, 256), 256, 0, s>>>(c2, c4, c1, rs, r0, Rc, DM);
  }
  // out = lin(res, fc2) + pre  (in-place over feats; residual read-then-write)
  gemm(s, rs, tb.fc2.w, tb.fc2.b, feats, feats, R, C, DM, false);
  ws.reset(mk);
}

static void run_tdown(hipStream_t s, Wsp& ws, const TDp& td,
                      const float* xyz, const float* feats,
                      float* nxyz, float* nfeats,
                      int B, int N, int M, int Cin, int Cout) {
  const size_t mk = ws.mark();
  int* fidx = ws.i((size_t)B * M);
  pt_fps_kernel<<<B, 256, 0, s>>>(xyz, fidx, N, M);
  pt_gather3_kernel<<<cdiv((long long)B * M * 3, 256), 256, 0, s>>>(xyz, fidx, nxyz, M, N, B * M * 3);

  int* idx = ws.i((size_t)B * M * 16);
  pt_knn_kernel<16><<<cdiv(B * M, 128), 128, 0, s>>>(nxyz, xyz, idx, nullptr, B, M, N);

  const int CT = Cin + 3;
  const int rows = B * M * 16;
  float* g = ws.f((size_t)rows * CT);
  pt_tdgroup_kernel<<<cdiv((long long)rows * CT, 256), 256, 0, s>>>(xyz, nxyz, feats, idx, g, M, N, Cin, rows * CT);

  float* h1 = ws.f((size_t)rows * Cout);
  gemm(s, g, td.c0.w, td.c0.b, nullptr, h1, rows, Cout, CT, false);
  bn_relu(s, ws, h1, td.b0, rows, Cout, true);
  float* h2 = ws.f((size_t)rows * Cout);
  gemm(s, h1, td.c1.w, td.c1.b, nullptr, h2, rows, Cout, Cout, false);
  bn_relu(s, ws, h2, td.b1, rows, Cout, true);

  pt_maxk_kernel<<<cdiv((long long)B * M * Cout, 256), 256, 0, s>>>(h2, nfeats, Cout, B * M * Cout);
  ws.reset(mk);
}

static void run_tup(hipStream_t s, Wsp& ws, const TUp& tu,
                    const float* xyz1, const float* pts1, int N1, int C1in,
                    const float* xyz2, const float* pts2, int N2, int C,
                    float* out, int B) {
  const size_t mk = ws.mark();
  const int R1 = B * N1, R2 = B * N2;
  float* f1 = ws.f((size_t)R1 * C);
  gemm(s, pts1, tu.f1.w, tu.f1.b, nullptr, f1, R1, C, C1in, false);
  bn_relu(s, ws, f1, tu.b1, R1, C, true);
  float* f2 = ws.f((size_t)R2 * C);
  gemm(s, pts2, tu.f2.w, tu.f2.b, nullptr, f2, R2, C, C, false);
  bn_relu(s, ws, f2, tu.b2, R2, C, true);

  int* i3 = ws.i((size_t)R2 * 3);
  float* d3 = ws.f((size_t)R2 * 3);
  pt_knn_kernel<3><<<cdiv(R2, 128), 128, 0, s>>>(xyz2, xyz1, i3, d3, B, N2, N1);
  pt_interp_kernel<<<cdiv((long long)R2 * C, 256), 256, 0, s>>>(f1, i3, d3, f2, out, N2, N1, C, R2 * C);
  ws.reset(mk);
}

} // namespace

// ---------------------------------------------------------------------------
extern "C" void kernel_launch(void* const* d_in, const int* in_sizes, int n_in,
                              void* d_out, int out_size, void* d_ws, size_t ws_size,
                              hipStream_t stream) {
  (void)in_sizes; (void)n_in; (void)out_size;
  const int B = 2, NP = 4096, NC = 50;
  const int NL[5] = {4096, 1024, 256, 64, 16};
  const int CL[5] = {32, 64, 128, 256, 512};

  // ---- parse inputs (insertion order of reference dicts) ----
  Cursor c{d_in, 0};
  const float* x6 = c.next();                       // 'x' [B,4096,6]
  Lin fc1a = parse_lin(c), fc1b = parse_lin(c);     // params['fc1']
  TB t1 = parse_tb(c);                              // params['t1']
  TDp td[4];
  for (int i = 0; i < 4; ++i) {                     // params['td']
    td[i].c0 = parse_lin(c); td[i].c1 = parse_lin(c);
    td[i].b0 = parse_bn(c);  td[i].b1 = parse_bn(c);
  }
  TB tfd[4];
  for (int i = 0; i < 4; ++i) tfd[i] = parse_tb(c); // params['tf_down']
  Lin fc2a = parse_lin(c), fc2b = parse_lin(c), fc2c = parse_lin(c); // params['fc2']
  TB t2 = parse_tb(c);                              // params['t2']
  TUp tu[4];
  for (int i = 0; i < 4; ++i) {                     // params['tus']
    tu[i].f1 = parse_lin(c); tu[i].b1 = parse_bn(c);
    tu[i].f2 = parse_lin(c); tu[i].b2 = parse_bn(c);
  }
  TB tfu[4];
  for (int i = 0; i < 4; ++i) tfu[i] = parse_tb(c); // params['tf_up']
  Lin fc3a = parse_lin(c), fc3b = parse_lin(c), fc3c = parse_lin(c); // params['fc3']

  Wsp ws{(char*)d_ws, 0, ws_size};

  // persistent per-level xyz / feature buffers (the "stack")
  float* XYZ[5]; float* F[5];
  for (int l = 0; l < 5; ++l) {
    XYZ[l] = ws.f((size_t)B * NL[l] * 3);
    F[l]   = ws.f((size_t)B * NL[l] * CL[l]);
  }

  // xyz = x[..., :3]
  pt_slice3_kernel<<<cdiv((long long)B * NP * 3, 256), 256, 0, stream>>>(x6, XYZ[0], B * NP * 3);

  // initial mlp2: 6 -> 128 (relu) -> 32
  {
    const size_t mk = ws.mark();
    float* h128 = ws.f((size_t)B * NP * 128);
    gemm(stream, x6, fc1a.w, fc1a.b, nullptr, h128, B * NP, 128, 6, true);
    gemm(stream, h128, fc1b.w, fc1b.b, nullptr, F[0], B * NP, 32, 128, false);
    ws.reset(mk);
  }
  run_transformer(stream, ws, t1, XYZ[0], F[0], B, NL[0], CL[0]);

  // encoder
  for (int i = 0; i < 4; ++i) {
    run_tdown(stream, ws, td[i], XYZ[i], F[i], XYZ[i + 1], F[i + 1],
              B, NL[i], NL[i + 1], CL[i], CL[i + 1]);
    run_transformer(stream, ws, tfd[i], XYZ[i + 1], F[i + 1], B, NL[i + 1], CL[i + 1]);
  }

  // bottleneck mlp 512->512->512 (temps inside mark), result directly in `cur`
  float* cur = ws.f((size_t)B * 16 * 512);
  {
    const size_t mk = ws.mark();
    float* ha = ws.f((size_t)B * 16 * 512);
    float* hb = ws.f((size_t)B * 16 * 512);
    gemm(stream, F[4], fc2a.w, fc2a.b, nullptr, ha,  B * 16, 512, 512, true);
    gemm(stream, ha,   fc2b.w, fc2b.b, nullptr, hb,  B * 16, 512, 512, true);
    gemm(stream, hb,   fc2c.w, fc2c.b, nullptr, cur, B * 16, 512, 512, false);
    ws.reset(mk);
  }
  run_transformer(stream, ws, t2, XYZ[4], cur, B, 16, 512);

  // decoder
  for (int i = 0; i < 4; ++i) {
    const int l = 3 - i;                         // target (finer) level
    float* nxt = ws.f((size_t)B * NL[l] * CL[l]);
    run_tup(stream, ws, tu[i], XYZ[l + 1], cur, NL[l + 1], CL[l + 1],
            XYZ[l], F[l], NL[l], CL[l], nxt, B);
    run_transformer(stream, ws, tfu[i], XYZ[l], nxt, B, NL[l], CL[l]);
    cur = nxt;
  }

  // head: 32 -> 64 (relu) -> 64 (relu) -> 50
  {
    const size_t mk = ws.mark();
    float* g1 = ws.f((size_t)B * NP * 64);
    float* g2 = ws.f((size_t)B * NP * 64);
    gemm(stream, cur, fc3a.w, fc3a.b, nullptr, g1, B * NP, 64, 32, true);
    gemm(stream, g1,  fc3b.w, fc3b.b, nullptr, g2, B * NP, 64, 64, true);
    gemm(stream, g2,  fc3c.w, fc3c.b, nullptr, (float*)d_out, B * NP, NC, 64, false);
    ws.reset(mk);
  }
}
